// GNNArmaTransformer_61194694034329
// MI455X (gfx1250) — matrix operand order repaired and checked by reading, not compiled, not run on hardware
//
#include <hip/hip_runtime.h>
#include <hip/hip_fp16.h>

typedef _Float16 half_t;
typedef __attribute__((ext_vector_type(16))) _Float16 v16h;
typedef __attribute__((ext_vector_type(8)))  _Float16 v8h;
typedef __attribute__((ext_vector_type(8)))  float    v8f;

#define WMMA_F16(a, b, c) \
  __builtin_amdgcn_wmma_f32_16x16x32_f16(false, (a), false, (b), (short)0, (c), false, false)

// ---------------------------------------------------------------------------
// Utility kernels
// ---------------------------------------------------------------------------
__global__ void zero_kernel(float* p, long long n) {
  long long i = (long long)blockIdx.x * blockDim.x + threadIdx.x;
  if (i < n) p[i] = 0.0f;
}

__global__ void cvt_kernel(const float* __restrict__ s, half_t* __restrict__ d, long long n) {
  long long i = (long long)blockIdx.x * blockDim.x + threadIdx.x;
  if (i < n) d[i] = (half_t)s[i];
}

// src [O][K] f32 -> dst [K][O] f16  (for torch-style W used as x @ W.T)
__global__ void cvtT_kernel(const float* __restrict__ s, half_t* __restrict__ d, int O, int Kd) {
  int i = blockIdx.x * blockDim.x + threadIdx.x;
  if (i >= O * Kd) return;
  int o = i / Kd, k = i % Kd;
  d[(long long)k * O + o] = (half_t)s[i];
}

// ---------------------------------------------------------------------------
// GCN norm
// ---------------------------------------------------------------------------
__global__ void deg_kernel(const long long* __restrict__ ei, float* deg, long long E) {
  long long e = (long long)blockIdx.x * blockDim.x + threadIdx.x;
  if (e < E) atomicAdd(&deg[(int)ei[E + e]], 1.0f);
}

__global__ void dis_kernel(float* deg, int n) {
  int i = blockIdx.x * blockDim.x + threadIdx.x;
  if (i < n) {
    float d = deg[i];
    deg[i] = d > 0.0f ? rsqrtf(fmaxf(d, 1.0f)) : 0.0f;
  }
}

__global__ void norm_kernel(const long long* __restrict__ ei, const float* __restrict__ dis,
                            float* __restrict__ nrm, long long E) {
  long long e = (long long)blockIdx.x * blockDim.x + threadIdx.x;
  if (e < E) nrm[e] = dis[(int)ei[e]] * dis[(int)ei[E + e]];
}

// agg[k][col] += norm[e] * src[k][row];  grid (E, K), block F
__global__ void scatter_kernel(const float* __restrict__ src, const long long* __restrict__ ei,
                               const float* __restrict__ nrm, float* __restrict__ agg,
                               int F, long long E, long long strideK) {
  long long e = blockIdx.x;
  int k = blockIdx.y, f = threadIdx.x;
  int row = (int)ei[e], col = (int)ei[E + e];
  float w = nrm[e];
  atomicAdd(&agg[(long long)k * strideK + (long long)col * F + f],
            w * src[(long long)k * strideK + (long long)row * F + f]);
}

__global__ void mean3_kernel(const float* __restrict__ a, float* __restrict__ o, long long S) {
  long long i = (long long)blockIdx.x * blockDim.x + threadIdx.x;
  if (i < S) o[i] = (a[i] + a[i + S] + a[i + 2 * S]) * (1.0f / 3.0f);
}

// ---------------------------------------------------------------------------
// WMMA GEMM: C[M,Nn] = epi(A[M,Kd] * B[Kd,Nn])  (A,B f16; M%128==0, Nn%64==0, Kd%32==0)
// Templated epilogue: no per-element control flow around the WMMA core.
// ---------------------------------------------------------------------------
template <bool HAS_ADD, bool HAS_BIAS, bool RELU, bool OUT16>
__global__ __launch_bounds__(256) void gemm16_kernel(
    const half_t* __restrict__ A, const half_t* __restrict__ B,
    const float* __restrict__ addsrc, const float* __restrict__ bias,
    float* __restrict__ Cf, half_t* __restrict__ Ch,
    int M, int Nn, int Kd) {
  __shared__ half_t As[128 * 40];
  __shared__ half_t Bs[64 * 40];
  const int tid = threadIdx.x;
  const int wave = tid >> 5, lane = tid & 31;
  const int hi = lane >> 4, lm = lane & 15;
  const int m0 = blockIdx.y * 128;
  const int n0 = blockIdx.x * 64;

  // staging indices (constant across K loop)
  const int srow = tid >> 1, sseg = tid & 1;       // A staging
  const int skk = tid & 31, sgrp = tid >> 5;       // B staging

  v8f acc[4];
#pragma unroll
  for (int j = 0; j < 4; ++j)
#pragma unroll
    for (int r = 0; r < 8; ++r) acc[j][r] = 0.0f;

  const half_t* gA = A + (long long)(m0 + srow) * Kd + sseg * 16;
  const half_t* gB = B + (long long)skk * Nn + n0 + sgrp * 8;

  for (int k0 = 0; k0 < Kd; k0 += 32) {
    // stage A: 128x32 tile, each thread moves 16 contiguous halfs
    {
      const half_t* gp = gA + k0;
      v8h a0 = *(const v8h*)gp;
      v8h a1 = *(const v8h*)(gp + 8);
      *(v8h*)&As[srow * 40 + sseg * 16] = a0;
      *(v8h*)&As[srow * 40 + sseg * 16 + 8] = a1;
      if (k0 + 32 < Kd) __builtin_prefetch(gp + 32, 0, 3);  // global_prefetch_b8
    }
    // stage B transposed: Bs[n][k]
    {
      const half_t* gp = gB + (long long)k0 * Nn;
      v8h b0 = *(const v8h*)gp;
#pragma unroll
      for (int i = 0; i < 8; ++i) Bs[(sgrp * 8 + i) * 40 + skk] = b0[i];
      if (k0 + 32 < Kd) __builtin_prefetch(gp + (long long)32 * Nn, 0, 3);
    }
    __syncthreads();

    // A fragment (contiguous k=[hi*16,+16), same k order as the B fragment)
    v16h afrag;
    {
      const half_t* ap = &As[(wave * 16 + lm) * 40 + hi * 16];
      *(v8h*)&afrag = *(const v8h*)ap;
      *((v8h*)&afrag + 1) = *(const v8h*)(ap + 8);
    }
#pragma unroll
    for (int j = 0; j < 4; ++j) {
      v16h bfrag;
      const half_t* bp = &Bs[(j * 16 + lm) * 40 + hi * 16];
      *(v8h*)&bfrag = *(const v8h*)bp;
      *((v8h*)&bfrag + 1) = *(const v8h*)(bp + 8);
      acc[j] = WMMA_F16(afrag, bfrag, acc[j]);
    }
    __syncthreads();
  }

  // epilogue: C layout — lane holds col n=lm, rows r+8*hi
  const int rowbase = m0 + wave * 16 + 8 * hi;
#pragma unroll
  for (int j = 0; j < 4; ++j) {
    const int col = n0 + j * 16 + lm;
    float bcol = 0.0f;
    if (HAS_BIAS) bcol = bias[col];
#pragma unroll
    for (int r = 0; r < 8; ++r) {
      const long long idx = (long long)(rowbase + r) * Nn + col;
      float v = acc[j][r];
      if (HAS_ADD) v += addsrc[idx];
      if (HAS_BIAS) v += bcol;
      if (RELU) v = fmaxf(v, 0.0f);
      if (OUT16) Ch[idx] = (half_t)v;
      else Cf[idx] = v;
    }
  }
}

// ---------------------------------------------------------------------------
// BatchNorm(+ReLU) in place: one block per feature column
// ---------------------------------------------------------------------------
__global__ __launch_bounds__(256) void bn_relu_kernel(float* __restrict__ h,
                                                      const float* __restrict__ g,
                                                      const float* __restrict__ be,
                                                      int N, int F) {
  int c = blockIdx.x;
  __shared__ float s1[256], s2[256];
  float a = 0.0f, b = 0.0f;
  for (int r = threadIdx.x; r < N; r += 256) {
    float v = h[(long long)r * F + c];
    a += v; b += v * v;
  }
  s1[threadIdx.x] = a; s2[threadIdx.x] = b;
  __syncthreads();
  for (int st = 128; st > 0; st >>= 1) {
    if (threadIdx.x < st) { s1[threadIdx.x] += s1[threadIdx.x + st]; s2[threadIdx.x] += s2[threadIdx.x + st]; }
    __syncthreads();
  }
  float mean = s1[0] / N;
  float var = s2[0] / N - mean * mean;
  float rs = rsqrtf(var + 1e-5f) * g[c];
  float bb = be[c];
  for (int r = threadIdx.x; r < N; r += 256) {
    float v = (h[(long long)r * F + c] - mean) * rs + bb;
    h[(long long)r * F + c] = fmaxf(v, 0.0f);
  }
}

// out = LayerNorm(a + b); one block (D threads) per row, D == 128
__global__ __launch_bounds__(128) void add_ln_kernel(const float* __restrict__ a,
                                                     const float* __restrict__ b,
                                                     const float* __restrict__ g,
                                                     const float* __restrict__ be,
                                                     float* __restrict__ out, int D) {
  int row = blockIdx.x, t = threadIdx.x;
  __shared__ float s1[128], s2[128];
  float v = a[(long long)row * D + t] + b[(long long)row * D + t];
  s1[t] = v; s2[t] = v * v;
  __syncthreads();
  for (int st = 64; st > 0; st >>= 1) {
    if (t < st) { s1[t] += s1[t + st]; s2[t] += s2[t + st]; }
    __syncthreads();
  }
  float mean = s1[0] / D;
  float var = s2[0] / D - mean * mean;
  out[(long long)row * D + t] = (v - mean) * rsqrtf(var + 1e-5f) * g[t] + be[t];
}

// vt[h*32+d][n] = qkv16[n][256 + h*32 + d]
__global__ void vt_kernel(const half_t* __restrict__ qkv, half_t* __restrict__ vt, int N) {
  long long i = (long long)blockIdx.x * blockDim.x + threadIdx.x;
  if (i >= (long long)N * 128) return;
  int n = (int)(i >> 7), c = (int)(i & 127);
  vt[(long long)c * N + n] = qkv[(long long)n * 384 + 256 + c];
}

// ---------------------------------------------------------------------------
// Flash attention: one wave per (16-query tile, head); dh = 32, D = 128
// ---------------------------------------------------------------------------
__global__ __launch_bounds__(32) void flash_attn_kernel(const half_t* __restrict__ qkv,
                                                        const half_t* __restrict__ vt,
                                                        half_t* __restrict__ o16, int N) {
  const int h = blockIdx.y;
  const int q0 = blockIdx.x * 16;
  const int lane = threadIdx.x & 31;
  const int hi = lane >> 4, lm = lane & 15;
  const float scale = 0.17677669529663688f;  // 1/sqrt(32)
  __shared__ half_t Pl[16 * 40];

  // Q fragment (A layout)
  v16h qa;
  {
    const half_t* qrow = qkv + (long long)(q0 + lm) * 384 + h * 32;
    *(v8h*)&qa = *(const v8h*)(qrow + hi * 16);
    *((v8h*)&qa + 1) = *(const v8h*)(qrow + hi * 16 + 8);
  }

  float mrun[8], lrun[8];
  v8f O0, O1;
#pragma unroll
  for (int r = 0; r < 8; ++r) { mrun[r] = -1e30f; lrun[r] = 0.0f; O0[r] = 0.0f; O1[r] = 0.0f; }

  for (int kv = 0; kv < N; kv += 32) {
    // K fragments (B layout): lane col = key index, elements = contiguous dh
    v16h kb0, kb1;
    {
      const half_t* kr0 = qkv + (long long)(kv + lm) * 384 + 128 + h * 32 + hi * 16;
      const half_t* kr1 = qkv + (long long)(kv + 16 + lm) * 384 + 128 + h * 32 + hi * 16;
      *(v8h*)&kb0 = *(const v8h*)kr0; *((v8h*)&kb0 + 1) = *(const v8h*)(kr0 + 8);
      *(v8h*)&kb1 = *(const v8h*)kr1; *((v8h*)&kb1 + 1) = *(const v8h*)(kr1 + 8);
    }
    v8f S0, S1;
#pragma unroll
    for (int r = 0; r < 8; ++r) { S0[r] = 0.0f; S1[r] = 0.0f; }
    S0 = WMMA_F16(qa, kb0, S0);
    S1 = WMMA_F16(qa, kb1, S1);

    // online softmax per row (rows spread: reg r, lane-half hi -> row r+8*hi)
#pragma unroll
    for (int r = 0; r < 8; ++r) {
      float s0 = S0[r] * scale, s1 = S1[r] * scale;
      float mx = fmaxf(s0, s1);
      for (int d = 1; d < 16; d <<= 1) mx = fmaxf(mx, __shfl_xor(mx, d, 32));
      float mnew = fmaxf(mrun[r], mx);
      float p0 = __expf(s0 - mnew), p1 = __expf(s1 - mnew);
      float ps = p0 + p1;
      for (int d = 1; d < 16; d <<= 1) ps += __shfl_xor(ps, d, 32);
      float alpha = __expf(mrun[r] - mnew);
      lrun[r] = lrun[r] * alpha + ps;
      mrun[r] = mnew;
      O0[r] *= alpha; O1[r] *= alpha;
      Pl[(r + 8 * hi) * 40 + lm] = (half_t)p0;
      Pl[(r + 8 * hi) * 40 + 16 + lm] = (half_t)p1;
    }
    __syncthreads();

    // P fragment (A layout) from LDS
    v16h pa;
    {
      const half_t* pp = &Pl[lm * 40 + hi * 16];
      *(v8h*)&pa = *(const v8h*)pp;
      *((v8h*)&pa + 1) = *(const v8h*)(pp + 8);
    }
    // V fragments from transposed V [dh][key]
    v16h vb0, vb1;
    {
      const half_t* v0p = vt + ((long long)h * 32 + lm) * N + kv + hi * 16;
      const half_t* v1p = vt + ((long long)h * 32 + 16 + lm) * N + kv + hi * 16;
      *(v8h*)&vb0 = *(const v8h*)v0p; *((v8h*)&vb0 + 1) = *(const v8h*)(v0p + 8);
      *(v8h*)&vb1 = *(const v8h*)v1p; *((v8h*)&vb1 + 1) = *(const v8h*)(v1p + 8);
    }
    O0 = WMMA_F16(pa, vb0, O0);
    O1 = WMMA_F16(pa, vb1, O1);
    __syncthreads();
  }

#pragma unroll
  for (int r = 0; r < 8; ++r) {
    int row = q0 + r + 8 * hi;
    float inv = 1.0f / lrun[r];
    o16[(long long)row * 128 + h * 32 + lm] = (half_t)(O0[r] * inv);
    o16[(long long)row * 128 + h * 32 + 16 + lm] = (half_t)(O1[r] * inv);
  }
}

// ---------------------------------------------------------------------------
// Classifier: out[n,c] = h[n,:] . w[c,:] + b[c]  (tiny, scalar)
// ---------------------------------------------------------------------------
__global__ void cls_kernel(const float* __restrict__ h, const float* __restrict__ w,
                           const float* __restrict__ b, float* __restrict__ out, int N, int D) {
  int i = blockIdx.x * blockDim.x + threadIdx.x;
  if (i >= N * 2) return;
  int n = i >> 1, c = i & 1;
  float s = b[c];
  for (int k = 0; k < D; ++k) s += h[(long long)n * D + k] * w[c * D + k];
  out[i] = s;
}

// ---------------------------------------------------------------------------
// Host orchestration
// ---------------------------------------------------------------------------
static inline char* carve(char*& p, size_t bytes) {
  char* r = p;
  p += (bytes + 255) & ~(size_t)255;
  return r;
}

extern "C" void kernel_launch(void* const* d_in, const int* in_sizes, int n_in,
                              void* d_out, int out_size, void* d_ws, size_t ws_size,
                              hipStream_t stream) {
  (void)in_sizes; (void)n_in; (void)out_size; (void)ws_size;
  const int N = 6144;
  const long long E = 98304;
  const int Fin = 128, Fh = 256, D = 128, DFF = 2048, K = 3, L = 2;

  const float* x       = (const float*)d_in[0];
  const long long* ei  = (const long long*)d_in[1];
  const float* c1_init = (const float*)d_in[2];
  const float* c1_wv   = (const float*)d_in[3];
  const float* c1_root = (const float*)d_in[4];
  const float* c1_b    = (const float*)d_in[5];
  const float* bn1_g   = (const float*)d_in[6];
  const float* bn1_bb  = (const float*)d_in[7];
  const float* c2_init = (const float*)d_in[8];
  const float* c2_wv   = (const float*)d_in[9];
  const float* c2_root = (const float*)d_in[10];
  const float* c2_b    = (const float*)d_in[11];
  const float* bn2_g   = (const float*)d_in[12];
  const float* bn2_bb  = (const float*)d_in[13];
  const float* wqkv    = (const float*)d_in[14];
  const float* bqkv    = (const float*)d_in[15];
  const float* wo      = (const float*)d_in[16];
  const float* bo      = (const float*)d_in[17];
  const float* ln1g    = (const float*)d_in[18];
  const float* ln1b    = (const float*)d_in[19];
  const float* w1      = (const float*)d_in[20];
  const float* b1      = (const float*)d_in[21];
  const float* w2      = (const float*)d_in[22];
  const float* b2      = (const float*)d_in[23];
  const float* ln2g    = (const float*)d_in[24];
  const float* ln2b    = (const float*)d_in[25];
  const float* clsw    = (const float*)d_in[26];
  const float* clsb    = (const float*)d_in[27];

  char* ws = (char*)d_ws;
  float*  deg    = (float*)carve(ws, (size_t)N * 4);
  float*  nrm    = (float*)carve(ws, (size_t)E * 4);
  half_t* x16    = (half_t*)carve(ws, (size_t)N * Fin * 2);
  half_t* c1i16  = (half_t*)carve(ws, (size_t)K * Fin * Fh * 2);
  half_t* c1w16  = (half_t*)carve(ws, (size_t)K * Fh * Fh * 2);
  half_t* c1r16  = (half_t*)carve(ws, (size_t)2 * K * Fin * Fh * 2);
  half_t* c2i16  = (half_t*)carve(ws, (size_t)K * Fh * D * 2);
  half_t* c2w16  = (half_t*)carve(ws, (size_t)K * D * D * 2);
  half_t* c2r16  = (half_t*)carve(ws, (size_t)2 * K * Fh * D * 2);
  half_t* wqkv16 = (half_t*)carve(ws, (size_t)L * D * 3 * D * 2);
  half_t* wo16   = (half_t*)carve(ws, (size_t)L * D * D * 2);
  half_t* w1T16  = (half_t*)carve(ws, (size_t)L * D * DFF * 2);
  half_t* w2T16  = (half_t*)carve(ws, (size_t)L * DFF * D * 2);
  float*  outk   = (float*)carve(ws, (size_t)K * N * Fh * 4);
  float*  aggk   = (float*)carve(ws, (size_t)K * N * Fh * 4);
  half_t* tmp16  = (half_t*)carve(ws, (size_t)K * N * Fh * 2);
  float*  h256   = (float*)carve(ws, (size_t)N * Fh * 4);
  half_t* h256_16= (half_t*)carve(ws, (size_t)N * Fh * 2);
  float*  hA     = (float*)carve(ws, (size_t)N * D * 4);
  float*  hB     = (float*)carve(ws, (size_t)N * D * 4);
  float*  proj   = (float*)carve(ws, (size_t)N * D * 4);
  half_t* hcur16 = (half_t*)carve(ws, (size_t)N * D * 2);
  half_t* qkv16  = (half_t*)carve(ws, (size_t)N * 3 * D * 2);
  half_t* vt16   = (half_t*)carve(ws, (size_t)N * D * 2);
  half_t* o16    = (half_t*)carve(ws, (size_t)N * D * 2);
  half_t* ffn16  = (half_t*)carve(ws, (size_t)N * DFF * 2);

  auto zero = [&](float* p, long long n) {
    zero_kernel<<<(unsigned)((n + 255) / 256), 256, 0, stream>>>(p, n);
  };
  auto cvt = [&](const float* s, half_t* d, long long n) {
    cvt_kernel<<<(unsigned)((n + 255) / 256), 256, 0, stream>>>(s, d, n);
  };
  auto cvtT = [&](const float* s, half_t* d, int O, int Kd) {
    cvtT_kernel<<<(O * Kd + 255) / 256, 256, 0, stream>>>(s, d, O, Kd);
  };
  // dispatch the template combos actually used
  auto gemm = [&](const half_t* A, const half_t* B, const float* add, const float* bias,
                  float* Cf, half_t* Ch, int Nn, int Kd, int relu) {
    dim3 g(Nn / 64, N / 128);
    if (add) {  // ARMA combine: add + bias + relu -> f32
      gemm16_kernel<true, true, true, false><<<g, 256, 0, stream>>>(A, B, add, bias, Cf, Ch, N, Nn, Kd);
    } else if (Ch) {
      if (relu)  // FFN up-proj: bias + relu -> f16
        gemm16_kernel<false, true, true, true><<<g, 256, 0, stream>>>(A, B, add, bias, Cf, Ch, N, Nn, Kd);
      else       // QKV: bias -> f16
        gemm16_kernel<false, true, false, true><<<g, 256, 0, stream>>>(A, B, add, bias, Cf, Ch, N, Nn, Kd);
    } else if (bias) {  // wo / w2: bias -> f32
      gemm16_kernel<false, true, false, false><<<g, 256, 0, stream>>>(A, B, add, bias, Cf, Ch, N, Nn, Kd);
    } else {            // plain transform -> f32
      gemm16_kernel<false, false, false, false><<<g, 256, 0, stream>>>(A, B, add, bias, Cf, Ch, N, Nn, Kd);
    }
  };

  // ---- gcn_norm ----
  zero(deg, N);
  deg_kernel<<<(unsigned)((E + 255) / 256), 256, 0, stream>>>(ei, deg, E);
  dis_kernel<<<(N + 255) / 256, 256, 0, stream>>>(deg, N);
  norm_kernel<<<(unsigned)((E + 255) / 256), 256, 0, stream>>>(ei, deg, nrm, E);

  // ---- weight conversions ----
  cvt(x, x16, (long long)N * Fin);
  cvt(c1_init, c1i16, (long long)K * Fin * Fh);
  cvt(c1_wv, c1w16, (long long)K * Fh * Fh);
  cvt(c1_root, c1r16, (long long)2 * K * Fin * Fh);
  cvt(c2_init, c2i16, (long long)K * Fh * D);
  cvt(c2_wv, c2w16, (long long)K * D * D);
  cvt(c2_root, c2r16, (long long)2 * K * Fh * D);
  for (int l = 0; l < L; ++l) {
    cvtT(wqkv + (long long)l * 3 * D * D, wqkv16 + (long long)l * D * 3 * D, 3 * D, D);
    cvtT(wo + (long long)l * D * D, wo16 + (long long)l * D * D, D, D);
    cvtT(w1 + (long long)l * DFF * D, w1T16 + (long long)l * D * DFF, DFF, D);
    cvtT(w2 + (long long)l * D * DFF, w2T16 + (long long)l * DFF * D, D, DFF);
  }

  // ---- ARMA conv helper (executed twice with different dims) ----
  auto arma = [&](const half_t* xin16, int FinL, int FoutL, const half_t* i16,
                  const half_t* w16, const half_t* r16, const float* bptr, float* hout) {
    long long SK = (long long)N * FoutL;
    // t = 0: transform
    for (int k = 0; k < K; ++k)
      gemm(xin16, i16 + (long long)k * FinL * FoutL, nullptr, nullptr,
           outk + k * SK, nullptr, FoutL, FinL, 0);
    zero(aggk, K * SK);
    scatter_kernel<<<dim3((unsigned)E, K), FoutL, 0, stream>>>(outk, ei, nrm, aggk, FoutL, E, SK);
    for (int k = 0; k < K; ++k)
      gemm(xin16, r16 + (long long)k * FinL * FoutL, aggk + k * SK,
           bptr + (long long)k * FoutL, outk + k * SK, nullptr, FoutL, FinL, 1);
    // t = 1: recurrent transform
    cvt(outk, tmp16, K * SK);
    for (int k = 0; k < K; ++k)
      gemm(tmp16 + k * SK, w16 + (long long)k * FoutL * FoutL, nullptr, nullptr,
           outk + k * SK, nullptr, FoutL, FoutL, 0);
    zero(aggk, K * SK);
    scatter_kernel<<<dim3((unsigned)E, K), FoutL, 0, stream>>>(outk, ei, nrm, aggk, FoutL, E, SK);
    for (int k = 0; k < K; ++k)
      gemm(xin16, r16 + (long long)(K + k) * FinL * FoutL, aggk + k * SK,
           bptr + (long long)(K + k) * FoutL, outk + k * SK, nullptr, FoutL, FinL, 1);
    mean3_kernel<<<(unsigned)((SK + 255) / 256), 256, 0, stream>>>(outk, hout, SK);
  };

  // ---- conv1 + bn1 + relu ----
  arma(x16, Fin, Fh, c1i16, c1w16, c1r16, c1_b, h256);
  bn_relu_kernel<<<Fh, 256, 0, stream>>>(h256, bn1_g, bn1_bb, N, Fh);

  // ---- conv2 + bn2 + relu ----
  cvt(h256, h256_16, (long long)N * Fh);
  arma(h256_16, Fh, D, c2i16, c2w16, c2r16, c2_b, hA);
  bn_relu_kernel<<<D, 256, 0, stream>>>(hA, bn2_g, bn2_bb, N, D);

  // ---- transformer layers ----
  float* cur = hA;
  float* nxt = hB;
  for (int l = 0; l < L; ++l) {
    cvt(cur, hcur16, (long long)N * D);
    // QKV projection (f16 output)
    gemm(hcur16, wqkv16 + (long long)l * D * 3 * D, nullptr, bqkv + (long long)l * 3 * D,
         nullptr, qkv16, 3 * D, D, 0);
    vt_kernel<<<(N * 128 + 255) / 256, 256, 0, stream>>>(qkv16, vt16, N);
    flash_attn_kernel<<<dim3(N / 16, 4), 32, 0, stream>>>(qkv16, vt16, o16, N);
    // output projection + residual + LN1
    gemm(o16, wo16 + (long long)l * D * D, nullptr, bo + (long long)l * D,
         proj, nullptr, D, D, 0);
    add_ln_kernel<<<N, D, 0, stream>>>(cur, proj, ln1g + (long long)l * D,
                                       ln1b + (long long)l * D, nxt, D);
    // FFN
    cvt(nxt, hcur16, (long long)N * D);
    gemm(hcur16, w1T16 + (long long)l * D * DFF, nullptr, b1 + (long long)l * DFF,
         nullptr, ffn16, DFF, D, 1);
    gemm(ffn16, w2T16 + (long long)l * DFF * D, nullptr, b2 + (long long)l * D,
         proj, nullptr, D, DFF, 0);
    add_ln_kernel<<<N, D, 0, stream>>>(nxt, proj, ln2g + (long long)l * D,
                                       ln2b + (long long)l * D, cur, D);
  }

  // ---- classifier ----
  cls_kernel<<<(N * 2 + 255) / 256, 256, 0, stream>>>(cur, clsw, clsb, (float*)d_out, N, D);
}
